// ModalityEnhancer_29618094474258
// MI455X (gfx1250) — matrix operand order repaired and checked by reading, not compiled, not run on hardware
//
#include <hip/hip_runtime.h>
#include <hip/hip_bf16.h>
#include <cstdint>

// ---- problem constants -------------------------------------------------
#define BDIM   8
#define LDIM   4096
#define DDIM   256
#define DSTATE 16
#define DCONV  4
#define DINNER 512
#define DTRANK 16
#define BL     (BDIM * LDIM)   // 32768 rows

// ---- vector typedefs ---------------------------------------------------
typedef __attribute__((ext_vector_type(16))) __bf16 bf16x16;
typedef __attribute__((ext_vector_type(8)))  __bf16 bf16x8;
typedef __attribute__((ext_vector_type(4)))  __bf16 bf16x4;
typedef __attribute__((ext_vector_type(8)))  float  f32x8;
typedef __attribute__((ext_vector_type(4)))  float  f32x4;

static __device__ __forceinline__ __bf16 f2bf(float f) {
  unsigned int u = __builtin_bit_cast(unsigned int, f);
  unsigned int r = (u + 0x7FFFu + ((u >> 16) & 1u)) >> 16;   // RNE
  unsigned short s = (unsigned short)r;
  return __builtin_bit_cast(__bf16, s);
}

// ---- WMMA GEMM: C[M,N] = f32(A[M,K]) @ f32(B[K,N]) via bf16 WMMA -------
// block tile 128x64, K-tile 64 (2 WMMA k-steps/stage), 256 thr = 8 waves
// All K and N in this problem are multiples of 4 -> staging guards are
// whole-vector-or-zero (no per-element exec-mask divergence).
// EPI: 0 none, 1 +bias, 2 +bias,GELU, 3 +bias,softplus, 4 +bias,+residual
#define BM 128
#define BN 64
#define BK 64
#define LDA_S 72   // bf16 elems per LDS A row (144B rows; 16B aligned)
#define LDB_S 72   // bf16 elems per LDS B^T row

template <int EPI>
__global__ __launch_bounds__(256) void gemm_bf16_wmma(
    const float* __restrict__ A, int lda,
    const float* __restrict__ Bw, int ldb,
    float* __restrict__ C, int ldc,
    const float* __restrict__ bias,
    const float* __restrict__ resid,
    int M, int N, int K) {
  __shared__ __bf16 As[BM * LDA_S];
  __shared__ __bf16 Bs[BN * LDB_S];   // stored transposed: [n][k]

  const int tid   = threadIdx.x;
  const int lane  = tid & 31;
  const int wid   = tid >> 5;
  const int waveM = wid & 3;          // 4 waves along M
  const int waveN = wid >> 2;         // 2 waves along N
  const int half  = lane >> 4;        // lane half select (WMMA layout)
  const int l16   = lane & 15;
  const int rowBase = blockIdx.y * BM;
  const int colBase = blockIdx.x * BN;

  // staging coordinates (hoisted)
  const int aR  = tid >> 4;             // +8 per iter chunk? no: f-based below
  const int bN  = tid & 63;             // B: fixed output column per thread
  const int bKq = (tid >> 6) << 2;      // B: starting k (step 16 per iter)

  f32x8 acc[2][2];
#pragma unroll
  for (int i = 0; i < 2; ++i)
#pragma unroll
    for (int j = 0; j < 2; ++j)
#pragma unroll
      for (int r = 0; r < 8; ++r) acc[i][j][r] = 0.0f;

  const int kTiles = (K + BK - 1) / BK;
  for (int kt = 0; kt < kTiles; ++kt) {
    const int k0 = kt * BK;

    // ---- prefetch next K-tile (global_prefetch_b8) --------------------
    if (kt + 1 < kTiles) {
      __builtin_prefetch(A + (size_t)(rowBase + (tid >> 1)) * lda + (k0 + BK),
                         0, 3);
      __builtin_prefetch(Bw + (size_t)(k0 + BK + (tid >> 2)) * ldb + colBase,
                         0, 3);
    }

    // ---- stage A tile (128x64 f32 -> bf16), 8 float4 per thread -------
#pragma unroll
    for (int i = 0; i < 8; ++i) {
      int f  = tid + i * 256;        // 0..2047 (float4 index)
      int r  = f >> 4;               // row 0..127
      int kc = (f & 15) << 2;        // k 0..60 step 4
      f32x4 v = {0.f, 0.f, 0.f, 0.f};
      if (k0 + kc + 3 < K)           // whole-vector guard (K % 4 == 0)
        v = *(const f32x4*)(A + (size_t)(rowBase + r) * lda + (k0 + kc));
      bf16x4 pk = {f2bf(v.x), f2bf(v.y), f2bf(v.z), f2bf(v.w)};
      *(bf16x4*)(&As[r * LDA_S + kc]) = pk;
    }
    // ---- stage B tile (64x64 -> bf16, transposed to [n][k]) -----------
    // each thread: one column n, 4 consecutive k (lane-coalesced reads,
    // contiguous 8B ds_store)
#pragma unroll
    for (int i = 0; i < 4; ++i) {
      int kc = bKq + i * 16;         // k 0..60 step 4
      float t0 = 0.f, t1 = 0.f, t2 = 0.f, t3 = 0.f;
      if (k0 + kc + 3 < K && colBase + bN < N) {
        const float* src = Bw + (size_t)(k0 + kc) * ldb + (colBase + bN);
        t0 = src[0];
        t1 = src[(size_t)ldb];
        t2 = src[(size_t)2 * ldb];
        t3 = src[(size_t)3 * ldb];
      }
      bf16x4 pk = {f2bf(t0), f2bf(t1), f2bf(t2), f2bf(t3)};
      *(bf16x4*)(&Bs[bN * LDB_S + kc]) = pk;
    }
    __syncthreads();

    // ---- 2 WMMA k-steps per stage -------------------------------------
#pragma unroll
    for (int kk = 0; kk < 2; ++kk) {
      const int ko = kk * 32;
      // A 16x32 frag: lane<16 -> row=l16, K {0..7,16..23}; lane>=16 ->
      // K {8..15,24..31} (relative to ko)
      bf16x16 af[2], bfg[2];
#pragma unroll
      for (int i = 0; i < 2; ++i) {
        const __bf16* ap =
            &As[(waveM * 32 + i * 16 + l16) * LDA_S + ko + half * 8];
        bf16x8 lo = *(const bf16x8*)ap;         // K = half*8 .. +7
        bf16x8 hi = *(const bf16x8*)(ap + 16);  // K = 16+half*8 .. +7
        af[i] = __builtin_shufflevector(lo, hi, 0, 1, 2, 3, 4, 5, 6, 7,
                                        8, 9, 10, 11, 12, 13, 14, 15);
      }
      // B 32x16 frag: lane<16 -> col=l16, K {0..7,16..23}; lane>=16 -> K {8..15,24..31}
#pragma unroll
      for (int j = 0; j < 2; ++j) {
        const __bf16* bp =
            &Bs[(waveN * 32 + j * 16 + l16) * LDB_S + ko + half * 8];
        bf16x8 lo = *(const bf16x8*)bp;
        bf16x8 hi = *(const bf16x8*)(bp + 16);
        bfg[j] = __builtin_shufflevector(lo, hi, 0, 1, 2, 3, 4, 5, 6, 7,
                                         8, 9, 10, 11, 12, 13, 14, 15);
      }
#pragma unroll
      for (int i = 0; i < 2; ++i)
#pragma unroll
        for (int j = 0; j < 2; ++j)
          acc[i][j] = __builtin_amdgcn_wmma_f32_16x16x32_bf16(
              false, af[i], false, bfg[j], (short)0, acc[i][j], false, false);
    }
    __syncthreads();
  }

  // ---- epilogue: C/D layout: VGPR r -> M = r + half*8, col = l16 ------
#pragma unroll
  for (int i = 0; i < 2; ++i) {
#pragma unroll
    for (int j = 0; j < 2; ++j) {
      int col = colBase + waveN * 32 + j * 16 + l16;
      if (col >= N) continue;
      int row0 = rowBase + waveM * 32 + i * 16 + half * 8;
      float bv = 0.f;
      if (EPI != 0) bv = bias[col];
#pragma unroll
      for (int r = 0; r < 8; ++r) {
        float v = acc[i][j][r] + bv;
        if (EPI == 2) v = 0.5f * v * (1.0f + erff(v * 0.70710678118654752f));
        if (EPI == 3) v = (v > 20.f) ? v : log1pf(__expf(v));
        size_t off = (size_t)(row0 + r) * ldc + col;
        if (EPI == 4) v += resid[off];
        C[off] = v;
      }
    }
  }
  (void)aR;
}

// ---- layer norm over D=256, one row per 256-thread block ---------------
__global__ __launch_bounds__(256) void ln_kernel(
    const float* __restrict__ in, const float* __restrict__ g,
    const float* __restrict__ b, float* __restrict__ out) {
  __shared__ float red[256];
  const int row = blockIdx.x;
  const int tid = threadIdx.x;
  float v = in[(size_t)row * DDIM + tid];
  red[tid] = v;
  __syncthreads();
  for (int s = 128; s > 0; s >>= 1) {
    if (tid < s) red[tid] += red[tid + s];
    __syncthreads();
  }
  float mu = red[0] * (1.0f / DDIM);
  __syncthreads();
  float d = v - mu;
  red[tid] = d * d;
  __syncthreads();
  for (int s = 128; s > 0; s >>= 1) {
    if (tid < s) red[tid] += red[tid + s];
    __syncthreads();
  }
  float inv = rsqrtf(red[0] * (1.0f / DDIM) + 1e-5f);
  out[(size_t)row * DDIM + tid] = d * inv * g[tid] + b[tid];
}

// ---- causal depthwise conv (width 4) + SiLU; dir selects time mapping --
__global__ __launch_bounds__(256) void conv_silu_kernel(
    const float* __restrict__ xz, const float* __restrict__ cw,
    const float* __restrict__ cb, float* __restrict__ xc, int dir) {
  int idx = blockIdx.x * 256 + threadIdx.x;   // B*L*DINNER
  int di = idx & (DINNER - 1);
  int bt = idx >> 9;
  int t = bt & (LDIM - 1);
  int b = bt >> 12;
  float acc = cb[di];
#pragma unroll
  for (int k = 0; k < DCONV; ++k) {
    int tt = t - (DCONV - 1) + k;
    if (tt >= 0) {
      int ts = dir ? (LDIM - 1 - tt) : tt;
      acc += xz[((size_t)(b * LDIM + ts)) * 1024 + di] * cw[di * DCONV + k];
    }
  }
  float s = 1.f / (1.f + __expf(-acc));
  xc[(size_t)bt * DINNER + di] = acc * s;
}

// ---- selective scan: one thread per (batch, channel); 16 states in VGPRs
__global__ __launch_bounds__(512) void scan_kernel(
    const float* __restrict__ dtp, const float* __restrict__ xdbl,
    const float* __restrict__ xc, const float* __restrict__ xz,
    const float* __restrict__ A_log, const float* __restrict__ D_skip,
    float* __restrict__ yout, int dir) {
  const int b = blockIdx.x;
  const int di = threadIdx.x;
  float Arow[DSTATE], h[DSTATE];
#pragma unroll
  for (int s = 0; s < DSTATE; ++s) {
    Arow[s] = -__expf(A_log[di * DSTATE + s]);
    h[s] = 0.f;
  }
  const float Dv = D_skip[di];
  for (int t = 0; t < LDIM; ++t) {
    size_t base = (size_t)b * LDIM + t;
    float dtv = dtp[base * DINNER + di];
    float xv = xc[base * DINNER + di];
    const f32x4* bc = (const f32x4*)(xdbl + base * 48 + DTRANK);
    float Bv[DSTATE], Cv[DSTATE];
#pragma unroll
    for (int q = 0; q < 4; ++q) {
      f32x4 vb = bc[q];
      f32x4 vc = bc[4 + q];
#pragma unroll
      for (int e = 0; e < 4; ++e) {
        Bv[4 * q + e] = vb[e];
        Cv[4 * q + e] = vc[e];
      }
    }
    float dx = dtv * xv;
    float y = 0.f;
#pragma unroll
    for (int s = 0; s < DSTATE; ++s) {
      float dA = __expf(dtv * Arow[s]);
      h[s] = dA * h[s] + dx * Bv[s];
      y += h[s] * Cv[s];
    }
    int tz = dir ? (LDIM - 1 - t) : t;
    float zv = xz[((size_t)b * LDIM + tz) * 1024 + DINNER + di];
    float sig = 1.f / (1.f + __expf(-zv));
    y = (y + Dv * xv) * (zv * sig);
    yout[base * DINNER + di] = y;
  }
}

// ---- x2 = x + 0.5*(y_fwd + flip(y_bwd)) --------------------------------
__global__ __launch_bounds__(256) void combine_kernel(
    const float* __restrict__ x, const float* __restrict__ yo0,
    const float* __restrict__ yo1, float* __restrict__ x2) {
  int idx = blockIdx.x * 256 + threadIdx.x;   // B*L*D
  int d = idx & (DDIM - 1);
  int bt = idx >> 8;
  int t = bt & (LDIM - 1);
  int b = bt >> 12;
  size_t idxr = ((size_t)(b * LDIM + (LDIM - 1 - t))) * DDIM + d;
  x2[idx] = x[idx] + 0.5f * (yo0[idx] + yo1[idxr]);
}

// ---- host-side orchestration -------------------------------------------
extern "C" void kernel_launch(void* const* d_in, const int* in_sizes, int n_in,
                              void* d_out, int out_size, void* d_ws,
                              size_t ws_size, hipStream_t stream) {
  const float* x      = (const float*)d_in[0];
  const float* ln1_g  = (const float*)d_in[1];
  const float* ln1_b  = (const float*)d_in[2];
  const float* W_in   = (const float*)d_in[3];
  const float* conv_w = (const float*)d_in[4];
  const float* conv_b = (const float*)d_in[5];
  const float* W_xprj = (const float*)d_in[6];
  const float* W_dtpj = (const float*)d_in[7];
  const float* dt_bias= (const float*)d_in[8];
  const float* A_log  = (const float*)d_in[9];
  const float* D_skip = (const float*)d_in[10];
  const float* W_out  = (const float*)d_in[11];
  const float* ln2_g  = (const float*)d_in[12];
  const float* ln2_b  = (const float*)d_in[13];
  const float* W1     = (const float*)d_in[14];
  const float* b1     = (const float*)d_in[15];
  const float* W2     = (const float*)d_in[16];
  const float* b2     = (const float*)d_in[17];
  float* out = (float*)d_out;

  float* ws = (float*)d_ws;
  size_t o = 0;
  float* h    = ws + o; o += (size_t)BL * DDIM;    // reused as h2
  float* xz   = ws + o; o += (size_t)BL * 1024;    // reused as FFN hidden g
  float* xc   = ws + o; o += (size_t)BL * DINNER;
  float* xdb  = ws + o; o += (size_t)BL * 48;
  float* dtb  = ws + o; o += (size_t)BL * DINNER;
  float* ybuf = ws + o; o += (size_t)BL * DINNER;
  float* yo0  = ws + o; o += (size_t)BL * DDIM;
  float* yo1  = ws + o; o += (size_t)BL * DDIM;
  float* x2   = ws + o; o += (size_t)BL * DDIM;
  float* g2   = xz;   // FFN hidden overlays xz (free after scans)
  float* h2   = h;    // ln2 output overlays h (free after first GEMM)

  const int gM = BL / BM;   // 256 M-tiles

  // h = LN1(x)
  ln_kernel<<<BL, 256, 0, stream>>>(x, ln1_g, ln1_b, h);
  // xz = h @ W_in   (direction-independent: pointwise in t)
  gemm_bf16_wmma<0><<<dim3(1024 / BN, gM), 256, 0, stream>>>(
      h, DDIM, W_in, 1024, xz, 1024, nullptr, nullptr, BL, 1024, DDIM);

  for (int dir = 0; dir < 2; ++dir) {
    // xc = silu(causal depthwise conv(xi, dir))
    conv_silu_kernel<<<(BL * DINNER) / 256, 256, 0, stream>>>(
        xz, conv_w, conv_b, xc, dir);
    // x_dbl = xc @ W_xproj  (N=48, one N-tile with bounds guard)
    gemm_bf16_wmma<0><<<dim3(1, gM), 256, 0, stream>>>(
        xc, DINNER, W_xprj, 48, xdb, 48, nullptr, nullptr, BL, 48, DINNER);
    // dt = softplus(x_dbl[:, :16] @ W_dtproj + dt_bias)  (K=16, zero-padded)
    gemm_bf16_wmma<3><<<dim3(DINNER / BN, gM), 256, 0, stream>>>(
        xdb, 48, W_dtpj, DINNER, dtb, DINNER, dt_bias, nullptr,
        BL, DINNER, DTRANK);
    // ybuf = scan(...) fused with D_skip add and silu(z) gate
    scan_kernel<<<BDIM, DINNER, 0, stream>>>(dtb, xdb, xc, xz, A_log, D_skip,
                                             ybuf, dir);
    // yo = ybuf @ W_out
    gemm_bf16_wmma<0><<<dim3(DDIM / BN, gM), 256, 0, stream>>>(
        ybuf, DINNER, W_out, DDIM, dir ? yo1 : yo0, DDIM, nullptr, nullptr,
        BL, DDIM, DINNER);
  }

  // x2 = x + 0.5*(yo0 + flip_t(yo1))
  combine_kernel<<<(BL * DDIM) / 256, 256, 0, stream>>>(x, yo0, yo1, x2);
  // h2 = LN2(x2)
  ln_kernel<<<BL, 256, 0, stream>>>(x2, ln2_g, ln2_b, h2);
  // g2 = gelu(h2 @ W1 + b1)
  gemm_bf16_wmma<2><<<dim3(1024 / BN, gM), 256, 0, stream>>>(
      h2, DDIM, W1, 1024, g2, 1024, b1, nullptr, BL, 1024, DDIM);
  // out = x2 + (g2 @ W2 + b2)
  gemm_bf16_wmma<4><<<dim3(DDIM / BN, gM), 256, 0, stream>>>(
      g2, 1024, W2, DDIM, out, DDIM, b2, x2, BL, DDIM, 1024);
}
